// RelationalAGG_9208409883351
// MI455X (gfx1250) — matrix operand order repaired and checked by reading, not compiled, not run on hardware
//
#include <hip/hip_runtime.h>
#include <hip/hip_bf16.h>

typedef __attribute__((ext_vector_type(16))) __bf16 v16bf;
typedef __attribute__((ext_vector_type(8)))  float  v8f;

#define DD   256
#define NNA  100000
#define NNB  100000
#define NE   500000
#define LN_EPS 1e-5f

// ---------------------------------------------------------------------------
// zero-fill (d_out is the aggregation accumulator; harness poisons with 0xAA)
// ---------------------------------------------------------------------------
__global__ void zero_kernel(float* __restrict__ p, long long n) {
  long long i = (long long)blockIdx.x * blockDim.x + threadIdx.x;
  long long stride = (long long)gridDim.x * blockDim.x;
  for (; i < n; i += stride) p[i] = 0.0f;
}

// ---------------------------------------------------------------------------
// C[M,256] = A[M,256] @ W^T  (W is [256,256] row-major, so B[k][n] = W[n*256+k])
// bf16 WMMA, f32 accumulate. One wave computes a 16x128 strip of C.
// A-frag layout (16-bit A 16x32, ISA 7.12.2): lane L holds row m = tm*16 + L%16,
//   VGPR v holds K pair k = 2*(v&3) + 8*(L/16) + 16*(v>>2).
// B-frag layout (16-bit B 32x16, same K order as A): lane L holds col
//   n = L%16, VGPR v holds K pair k = 2*(v&3) + 8*(L/16) + 16*(v>>2).
// Grid is sized exactly (M % 16 == 0) so EXEC is all-ones for WMMA.
// ---------------------------------------------------------------------------
__global__ __launch_bounds__(128) void gemm_bf16_wmma(
    const float* __restrict__ A,   // [M, 256] f32
    const float* __restrict__ W,   // [256, 256] f32 row-major
    float* __restrict__ C,         // [M, 256] f32
    int M) {
  const int lane  = threadIdx.x & 31;
  const int wid   = threadIdx.x >> 5;
  const int gw    = blockIdx.x * 4 + wid;   // global wave id
  const int strip = gw & 1;                 // which 128-col half
  const int tm    = gw >> 1;                // 16-row tile index
  const int l16   = lane & 15;
  const int half  = lane >> 4;
  const int row   = tm * 16 + l16;
  const int colb  = strip * 128 + l16;

  v8f acc[8];
#pragma unroll
  for (int t = 0; t < 8; ++t) acc[t] = (v8f){0,0,0,0,0,0,0,0};

  const float* arow = A + (long long)row * DD;

  for (int k0 = 0; k0 < DD; k0 += 32) {
    v16bf af;
#pragma unroll
    for (int v = 0; v < 8; ++v) {
      const int k = k0 + 2 * (v & 3) + 8 * half + 16 * (v >> 2);
      const float2 a2 = *(const float2*)(arow + k);
      af[2 * v]     = (__bf16)a2.x;
      af[2 * v + 1] = (__bf16)a2.y;
    }
#pragma unroll
    for (int t = 0; t < 8; ++t) {
      const float* wrow = W + (long long)(colb + t * 16) * DD;  // B column = W row
      v16bf bfr;
#pragma unroll
      for (int v = 0; v < 8; ++v) {
        const int k = k0 + 2 * (v & 3) + 8 * half + 16 * (v >> 2);
        const float2 b2 = *(const float2*)(wrow + k);
        bfr[2 * v]     = (__bf16)b2.x;
        bfr[2 * v + 1] = (__bf16)b2.y;
      }
      acc[t] = __builtin_amdgcn_wmma_f32_16x16x32_bf16(
          false, af, false, bfr, (short)0, acc[t], false, false);
    }
  }

  // D/C layout: VGPR r of lane L is element (m = tm*16 + r + 8*(L/16), n = colb)
#pragma unroll
  for (int t = 0; t < 8; ++t) {
#pragma unroll
    for (int r = 0; r < 8; ++r) {
      const int m = tm * 16 + r + 8 * half;
      C[(long long)m * DD + colb + t * 16] = acc[t][r];
    }
  }
}

// ---------------------------------------------------------------------------
// One wave per edge:
//   s = exp( sum_j t_h[src][j] * h_dst[dst][j] * WA[j] )
//   att_sum[dst]   += s                (atomic)
//   out_acc[dst,:] += s * h_src[src,:] (atomic; normalization deferred)
// ---------------------------------------------------------------------------
__global__ __launch_bounds__(256) void edge_kernel(
    const float* __restrict__ t_h,   // [Nsrc, 256]
    const float* __restrict__ hdst,  // [Ndst, 256]
    const float* __restrict__ hsrc,  // [Nsrc, 256]
    const int*   __restrict__ src,
    const int*   __restrict__ dst,
    const float* __restrict__ WA,    // [256]
    float* __restrict__ att_sum,     // [Ndst]
    float* __restrict__ out_acc,     // [Ndst, 256]
    int E) {
  const int lane = threadIdx.x & 31;
  const int e    = blockIdx.x * 8 + (threadIdx.x >> 5);
  if (e >= E) return;

  const int s = src[e];
  const int d = dst[e];
  const float* ts = t_h + (long long)s * DD;
  const float* hd = hdst + (long long)d * DD;
  const float* hs = hsrc + (long long)s * DD;

  const int j0 = lane * 4;
  const int j1 = 128 + lane * 4;

  const float4 t0 = *(const float4*)(ts + j0);
  const float4 t1 = *(const float4*)(ts + j1);
  const float4 v0 = *(const float4*)(hd + j0);
  const float4 v1 = *(const float4*)(hd + j1);
  const float4 w0 = *(const float4*)(WA + j0);
  const float4 w1 = *(const float4*)(WA + j1);

  float partial = t0.x * v0.x * w0.x + t0.y * v0.y * w0.y +
                  t0.z * v0.z * w0.z + t0.w * v0.w * w0.w +
                  t1.x * v1.x * w1.x + t1.y * v1.y * w1.y +
                  t1.z * v1.z * w1.z + t1.w * v1.w * w1.w;

#pragma unroll
  for (int off = 16; off > 0; off >>= 1)
    partial += __shfl_xor(partial, off, 32);

  const float sE = __expf(partial);
  if (lane == 0) atomicAdd(att_sum + d, sE);

  const float4 s0 = *(const float4*)(hs + j0);
  const float4 s1 = *(const float4*)(hs + j1);
  float* p = out_acc + (long long)d * DD;
  atomicAdd(p + j0 + 0, sE * s0.x);
  atomicAdd(p + j0 + 1, sE * s0.y);
  atomicAdd(p + j0 + 2, sE * s0.z);
  atomicAdd(p + j0 + 3, sE * s0.w);
  atomicAdd(p + j1 + 0, sE * s1.x);
  atomicAdd(p + j1 + 1, sE * s1.y);
  atomicAdd(p + j1 + 2, sE * s1.z);
  atomicAdd(p + j1 + 3, sE * s1.w);
}

// ---------------------------------------------------------------------------
// One wave per node: h = relu(acc / att_sum); LayerNorm(h)*gamma + beta.
// In place on d_out. att_sum == 0 (no in-edges) -> h = 0 -> output = beta,
// matching the reference's segment_sum-of-empty behavior.
// ---------------------------------------------------------------------------
__global__ __launch_bounds__(256) void finalize_kernel(
    float* __restrict__ out,          // [N, 256] in/out
    const float* __restrict__ att_sum,
    const float* __restrict__ gamma,
    const float* __restrict__ beta,
    int N) {
  const int lane = threadIdx.x & 31;
  const int n    = blockIdx.x * 8 + (threadIdx.x >> 5);
  if (n >= N) return;

  const float asum = att_sum[n];
  const float rcp  = asum > 0.0f ? 1.0f / asum : 0.0f;
  float* rowp = out + (long long)n * DD;

  const int j0 = lane * 4;
  const int j1 = 128 + lane * 4;
  const float4 a = *(const float4*)(rowp + j0);
  const float4 b = *(const float4*)(rowp + j1);

  float h[8] = {a.x, a.y, a.z, a.w, b.x, b.y, b.z, b.w};
  float lsum = 0.0f;
#pragma unroll
  for (int i = 0; i < 8; ++i) {
    h[i] = fmaxf(h[i] * rcp, 0.0f);
    lsum += h[i];
  }
#pragma unroll
  for (int off = 16; off > 0; off >>= 1) lsum += __shfl_xor(lsum, off, 32);
  const float mu = lsum * (1.0f / 256.0f);

  float lvar = 0.0f;
#pragma unroll
  for (int i = 0; i < 8; ++i) {
    const float dv = h[i] - mu;
    lvar += dv * dv;
  }
#pragma unroll
  for (int off = 16; off > 0; off >>= 1) lvar += __shfl_xor(lvar, off, 32);
  const float inv = rsqrtf(lvar * (1.0f / 256.0f) + LN_EPS);

  const float4 g0 = *(const float4*)(gamma + j0);
  const float4 g1 = *(const float4*)(gamma + j1);
  const float4 b0 = *(const float4*)(beta + j0);
  const float4 b1 = *(const float4*)(beta + j1);

  float4 o0, o1;
  o0.x = (h[0] - mu) * inv * g0.x + b0.x;
  o0.y = (h[1] - mu) * inv * g0.y + b0.y;
  o0.z = (h[2] - mu) * inv * g0.z + b0.z;
  o0.w = (h[3] - mu) * inv * g0.w + b0.w;
  o1.x = (h[4] - mu) * inv * g1.x + b1.x;
  o1.y = (h[5] - mu) * inv * g1.y + b1.y;
  o1.z = (h[6] - mu) * inv * g1.z + b1.z;
  o1.w = (h[7] - mu) * inv * g1.w + b1.w;
  *(float4*)(rowp + j0) = o0;
  *(float4*)(rowp + j1) = o1;
}

// ---------------------------------------------------------------------------
extern "C" void kernel_launch(void* const* d_in, const int* in_sizes, int n_in,
                              void* d_out, int out_size, void* d_ws, size_t ws_size,
                              hipStream_t stream) {
  const float* feat_a = (const float*)d_in[0];
  const float* feat_b = (const float*)d_in[1];
  const int*   src_ab = (const int*)d_in[2];
  const int*   dst_ab = (const int*)d_in[3];
  const int*   src_ba = (const int*)d_in[4];
  const int*   dst_ba = (const int*)d_in[5];
  const float* WT_ab  = (const float*)d_in[6];
  const float* WT_ba  = (const float*)d_in[7];
  const float* WA_ab  = (const float*)d_in[8];
  const float* WA_ba  = (const float*)d_in[9];
  const float* gamma  = (const float*)d_in[10];
  const float* beta   = (const float*)d_in[11];

  float* out   = (float*)d_out;
  float* out_a = out;                       // [NA, D]  (stack index 0)
  float* out_b = out + (size_t)NNA * DD;    // [NB, D]  (stack index 1)

  float* ws      = (float*)d_ws;
  float* t_h_ab  = ws;                              // [NA, D]
  float* t_h_ba  = t_h_ab + (size_t)NNA * DD;       // [NB, D]
  float* att_b   = t_h_ba + (size_t)NNB * DD;       // [NB]
  float* att_a   = att_b + NNB;                     // [NA]

  // 1) zero accumulators (d_out) and attention sums
  zero_kernel<<<2048, 256, 0, stream>>>(out, (long long)(NNA + NNB) * DD);
  zero_kernel<<<256, 256, 0, stream>>>(att_b, (long long)(NNA + NNB));

  // 2) per-etype linear transforms via bf16 WMMA
  //    waves = (M/16)*2 strips; 4 waves/block -> M/32 blocks (M % 32 == 0)
  gemm_bf16_wmma<<<NNA / 32, 128, 0, stream>>>(feat_a, WT_ab, t_h_ab, NNA);
  gemm_bf16_wmma<<<NNB / 32, 128, 0, stream>>>(feat_b, WT_ba, t_h_ba, NNB);

  // 3) fused edge pass: s = exp(WA . (t_h[src] * h_dst[dst])),
  //    scatter s and s*h_src with atomics (normalization deferred)
  edge_kernel<<<NE / 8, 256, 0, stream>>>(t_h_ab, feat_b, feat_a,
                                          src_ab, dst_ab, WA_ab,
                                          att_b, out_b, NE);
  edge_kernel<<<NE / 8, 256, 0, stream>>>(t_h_ba, feat_a, feat_b,
                                          src_ba, dst_ba, WA_ba,
                                          att_a, out_a, NE);

  // 4) divide by att_sum + ReLU + LayerNorm, in place in d_out
  finalize_kernel<<<NNA / 8, 256, 0, stream>>>(out_a, att_a, gamma, beta, NNA);
  finalize_kernel<<<NNB / 8, 256, 0, stream>>>(out_b, att_b, gamma, beta, NNB);
}
